// FullStateNonlinearityKAN_pyKAN_82377472737774
// MI455X (gfx1250) — matrix operand order repaired and checked by reading, not compiled, not run on hardware
//
#include <hip/hip_runtime.h>
#include <hip/hip_bf16.h>
#include <cstdint>

// ---------------------------------------------------------------------------
// KAN (pyKAN) 2-layer forward, restructured as feature-expansion + bf16 WMMA
// GEMM per layer.  B=2048, IN=128, HID=512, OUT=128, nb=G+K=8.
// Feature layout (k-major for coalesced stores): f = k*CH + i,
//   k=0..7 spline basis, k=8 silu(x_i);  Ktot = 9*CH.
// Weights folded to match: Wt[o, k*CH+i] = coef[i,o,k]*scale_sp[i,o],
//   Wt[o, 8*CH+i] = scale_base[i,o].   (Wt transposed: [N, Ktot])
// ---------------------------------------------------------------------------

typedef __attribute__((ext_vector_type(16))) __bf16 v16bf;
typedef __attribute__((ext_vector_type(8)))  __bf16 bf16x8;
typedef __attribute__((ext_vector_type(8)))  float  v8f;

union FragAB { v16bf v; bf16x8 h[2]; };

#define KAN_B    2048
#define KAN_IN   128
#define KAN_HID  512
#define KAN_OUT  128
#define KAN_NB   8      // G+K spline basis count
#define KAN_NF   9      // 8 basis + 1 silu
#define KAN_NK   12     // G+2K+1 knots per channel

// ---------------------------------------------------------------------------
// Weight fold: coef [CH, OUT, 8], scale_base/scale_sp [CH, OUT]
//   -> Wt bf16 [OUT, CH*9] with k-major feature index (k*CH + i).
// ---------------------------------------------------------------------------
__global__ void kan_fold_weights(const float* __restrict__ coef,
                                 const float* __restrict__ scale_base,
                                 const float* __restrict__ scale_sp,
                                 __bf16* __restrict__ Wt,
                                 int CH, int OUT) {
    int idx = blockIdx.x * blockDim.x + threadIdx.x;
    if (idx >= CH * OUT) return;
    int i = idx / OUT;
    int o = idx % OUT;
    float sp = scale_sp[(size_t)i * OUT + o];
    const float* c = coef + ((size_t)i * OUT + o) * KAN_NB;
    __bf16* w = Wt + (size_t)o * (CH * KAN_NF);
#pragma unroll
    for (int k = 0; k < KAN_NB; ++k)
        w[(size_t)k * CH + i] = (__bf16)(c[k] * sp);
    w[(size_t)KAN_NB * CH + i] = (__bf16)scale_base[(size_t)i * OUT + o];
}

// ---------------------------------------------------------------------------
// Feature expansion: x [B, CH] f32, grid [CH, 12] f32 -> F bf16 [B, CH*9]
// (k-major). Cox–de Boor recursion for cubic B-splines, 8 surviving bases.
// Consecutive lanes handle consecutive i -> all 9 stores fully coalesced.
// ---------------------------------------------------------------------------
__global__ void kan_features(const float* __restrict__ X,
                             const float* __restrict__ grid,
                             __bf16* __restrict__ F,
                             int Bn, int CH) {
    int idx = blockIdx.x * blockDim.x + threadIdx.x;
    if (idx >= Bn * CH) return;
    int b = idx / CH;
    int i = idx % CH;
    float x = X[idx];

    float t[KAN_NK];
    const float* g = grid + (size_t)i * KAN_NK;
#pragma unroll
    for (int j = 0; j < KAN_NK; ++j) t[j] = g[j];

    float Bv[KAN_NK - 1];
#pragma unroll
    for (int j = 0; j < KAN_NK - 1; ++j)
        Bv[j] = (x >= t[j] && x < t[j + 1]) ? 1.0f : 0.0f;

#pragma unroll
    for (int p = 1; p <= 3; ++p) {
#pragma unroll
        for (int j = 0; j < KAN_NK - 1 - p; ++j) {
            float a = (x - t[j])         / (t[j + p]     - t[j]);
            float c = (t[j + p + 1] - x) / (t[j + p + 1] - t[j + 1]);
            Bv[j] = a * Bv[j] + c * Bv[j + 1];
        }
    }

    __bf16* out = F + (size_t)b * (CH * KAN_NF) + i;
#pragma unroll
    for (int k = 0; k < KAN_NB; ++k)
        out[(size_t)k * CH] = (__bf16)Bv[k];
    float s = x / (1.0f + __expf(-x));   // silu
    out[(size_t)KAN_NB * CH] = (__bf16)s;
}

// ---------------------------------------------------------------------------
// WMMA GEMM:  C[M,N] f32 = A[M,K] bf16 (row-major) x Bt[N,K] bf16 (B transp.)
// One wave computes a 32(M) x 64(N) tile: 2 A-frags x 4 B-frags -> 8 WMMAs
// per K-step from 12 b128 loads (~21 FLOP/B from L2).
//
// M,N,K are template constants so all 12 loads and all 64 stores address as
// base + 24-bit immediate: exactly two live base pointers, bumped 64B/iter.
// (Removes the per-iteration u64 address chains + scratch spill seen when
//  K was a runtime value.)
//
// Fragment packing per CDNA5 ISA §7.12.2 (wave32):
//  A 16x32: lane l<16 -> row l, K = {0..7} then {16..23}; lane>=16 -> row l-16,
//           K = {8..15} then {24..31}   => two contiguous 16B loads.
//  B 32x16: lane l<16 -> col l, K = {0..7} then {16..23}; lane>=16 -> col l-16,
//           K = {8..15} then {24..31} => two 16B loads from the transposed weight matrix.
//  C 16x16: lane l -> col l&15, VGPR r -> row (l<16 ? r : 8+r).
// ---------------------------------------------------------------------------
template <int M, int N, int K>
__global__ __launch_bounds__(256, 1)
void kan_wmma_gemm(const __bf16* __restrict__ A,
                   const __bf16* __restrict__ Bt,
                   float* __restrict__ C) {
    constexpr int TILES_N = N / 64;            // 64-wide N super-tiles
    const int lane = threadIdx.x & 31;
    const int wave = (blockIdx.x * blockDim.x + threadIdx.x) >> 5;
    const int tm = wave / TILES_N;             // 32-row M tile
    const int tn = wave % TILES_N;

    const int rowA0 = tm * 32 + (lane & 15);
    const int kOffA = (lane < 16) ? 0 : 8;     // A half-select
    const int kOffB = (lane < 16) ? 0 : 8;     // B half-select
    const int colB  = tn * 64 + (lane & 15);

    v8f acc00 = {}, acc01 = {}, acc02 = {}, acc03 = {};
    v8f acc10 = {}, acc11 = {}, acc12 = {}, acc13 = {};

    // Single A base; second M-tile row at immediate offset 16*K elements.
    const __bf16* aBase = A + (size_t)rowA0 * K + kOffA;
    // Single B base; N-tiles j=0..3 at immediate offsets j*16*K elements.
    const __bf16* bBase = Bt + (size_t)colB * K + kOffB;

#pragma unroll 2
    for (int k0 = 0; k0 < K; k0 += 32) {
        FragAB a0, a1;
        a0.h[0] = *(const bf16x8*)(aBase);
        a0.h[1] = *(const bf16x8*)(aBase + 16);
        a1.h[0] = *(const bf16x8*)(aBase + 16 * K);
        a1.h[1] = *(const bf16x8*)(aBase + 16 * K + 16);

        FragAB b0, b1, b2, b3;
        b0.h[0] = *(const bf16x8*)(bBase);
        b0.h[1] = *(const bf16x8*)(bBase + 16);
        b1.h[0] = *(const bf16x8*)(bBase + 16 * K);
        b1.h[1] = *(const bf16x8*)(bBase + 16 * K + 16);
        b2.h[0] = *(const bf16x8*)(bBase + 32 * K);
        b2.h[1] = *(const bf16x8*)(bBase + 32 * K + 16);
        b3.h[0] = *(const bf16x8*)(bBase + 48 * K);
        b3.h[1] = *(const bf16x8*)(bBase + 48 * K + 16);

        aBase += 32;
        bBase += 32;

        acc00 = __builtin_amdgcn_wmma_f32_16x16x32_bf16(false, a0.v, false, b0.v,
                                                        (short)0, acc00, false, false);
        acc01 = __builtin_amdgcn_wmma_f32_16x16x32_bf16(false, a0.v, false, b1.v,
                                                        (short)0, acc01, false, false);
        acc02 = __builtin_amdgcn_wmma_f32_16x16x32_bf16(false, a0.v, false, b2.v,
                                                        (short)0, acc02, false, false);
        acc03 = __builtin_amdgcn_wmma_f32_16x16x32_bf16(false, a0.v, false, b3.v,
                                                        (short)0, acc03, false, false);
        acc10 = __builtin_amdgcn_wmma_f32_16x16x32_bf16(false, a1.v, false, b0.v,
                                                        (short)0, acc10, false, false);
        acc11 = __builtin_amdgcn_wmma_f32_16x16x32_bf16(false, a1.v, false, b1.v,
                                                        (short)0, acc11, false, false);
        acc12 = __builtin_amdgcn_wmma_f32_16x16x32_bf16(false, a1.v, false, b2.v,
                                                        (short)0, acc12, false, false);
        acc13 = __builtin_amdgcn_wmma_f32_16x16x32_bf16(false, a1.v, false, b3.v,
                                                        (short)0, acc13, false, false);
    }

    const int col   = lane & 15;
    const int rbase = (lane < 16) ? 0 : 8;
    float* c0 = C + (size_t)(tm * 32 + rbase) * N + tn * 64 + col;
#pragma unroll
    for (int r = 0; r < 8; ++r) {
        c0[r * N +  0] = acc00[r];
        c0[r * N + 16] = acc01[r];
        c0[r * N + 32] = acc02[r];
        c0[r * N + 48] = acc03[r];
        c0[(16 + r) * N +  0] = acc10[r];
        c0[(16 + r) * N + 16] = acc11[r];
        c0[(16 + r) * N + 32] = acc12[r];
        c0[(16 + r) * N + 48] = acc13[r];
    }
}

// ---------------------------------------------------------------------------
// Launch
// ---------------------------------------------------------------------------
extern "C" void kernel_launch(void* const* d_in, const int* in_sizes, int n_in,
                              void* d_out, int out_size, void* d_ws, size_t ws_size,
                              hipStream_t stream) {
    (void)in_sizes; (void)n_in; (void)out_size; (void)ws_size;

    const float* state       = (const float*)d_in[0];
    const float* grid1       = (const float*)d_in[1];
    const float* coef1       = (const float*)d_in[2];
    const float* scale_base1 = (const float*)d_in[3];
    const float* scale_sp1   = (const float*)d_in[4];
    const float* grid2       = (const float*)d_in[5];
    const float* coef2       = (const float*)d_in[6];
    const float* scale_base2 = (const float*)d_in[7];
    const float* scale_sp2   = (const float*)d_in[8];
    float* out = (float*)d_out;

    constexpr int K1 = KAN_IN  * KAN_NF;   // 1152
    constexpr int K2 = KAN_HID * KAN_NF;   // 4608

    // Workspace carve-up (all offsets multiples of 256B).
    uint8_t* ws = (uint8_t*)d_ws;
    __bf16* Wt1 = (__bf16*)(ws);                                   // 512*1152*2  = 1,179,648
    __bf16* F1  = (__bf16*)(ws + 1179648);                         // 2048*1152*2 = 4,718,592
    float*  H   = (float*) (ws + 1179648 + 4718592);               // 2048*512*4  = 4,194,304
    __bf16* Wt2 = (__bf16*)(ws + 1179648 + 4718592 + 4194304);     // 128*4608*2  = 1,179,648
    __bf16* F2  = (__bf16*)(ws + 1179648 + 4718592 + 4194304 + 1179648); // 2048*4608*2

    const int TPB = 256;

    // ---- Layer 1 ----
    kan_fold_weights<<<(KAN_IN * KAN_HID + TPB - 1) / TPB, TPB, 0, stream>>>(
        coef1, scale_base1, scale_sp1, Wt1, KAN_IN, KAN_HID);
    kan_features<<<(KAN_B * KAN_IN + TPB - 1) / TPB, TPB, 0, stream>>>(
        state, grid1, F1, KAN_B, KAN_IN);
    {   // (2048/32) * (512/64) = 512 waves; 8 waves/block -> 64 blocks
        int blocks = (KAN_B / 32) * (KAN_HID / 64) / (TPB / 32);
        kan_wmma_gemm<KAN_B, KAN_HID, K1><<<blocks, TPB, 0, stream>>>(F1, Wt1, H);
    }

    // ---- Layer 2 ----
    kan_fold_weights<<<(KAN_HID * KAN_OUT + TPB - 1) / TPB, TPB, 0, stream>>>(
        coef2, scale_base2, scale_sp2, Wt2, KAN_HID, KAN_OUT);
    kan_features<<<(KAN_B * KAN_HID + TPB - 1) / TPB, TPB, 0, stream>>>(
        H, grid2, F2, KAN_B, KAN_HID);
    {   // (2048/32) * (128/64) = 128 waves -> 16 blocks
        int blocks = (KAN_B / 32) * (KAN_OUT / 64) / (TPB / 32);
        kan_wmma_gemm<KAN_B, KAN_OUT, K2><<<blocks, TPB, 0, stream>>>(F2, Wt2, out);
    }
}
